// NonLocalBlock_10806137717502
// MI455X (gfx1250) — matrix operand order repaired and checked by reading, not compiled, not run on hardware
//
#include <hip/hip_runtime.h>
#include <hip/hip_fp16.h>

typedef _Float16 v16h __attribute__((ext_vector_type(16)));
typedef _Float16 v8h  __attribute__((ext_vector_type(8)));
typedef float    v8f  __attribute__((ext_vector_type(8)));

#define BATCH  8
#define CH     256
#define NPIX   4096
#define GROUPS 32
#define CPG    8

__device__ __forceinline__ int lane_id() { return (int)(threadIdx.x & 31); }

__device__ __forceinline__ v8f zero8() {
    v8f z;
#pragma unroll
    for (int i = 0; i < 8; ++i) z[i] = 0.0f;
    return z;
}

__device__ __forceinline__ v16h make16(v8h a, v8h b) {
    v16h r;
#pragma unroll
    for (int i = 0; i < 8; ++i) { r[i] = a[i]; r[i + 8] = b[i]; }
    return r;
}

// A-operand (16x32 f16, M x K), row-major source with row stride `stride` halves.
// ISA layout: lane L holds row M = L&15; half e holds K = (e<8 ? e : e+8) + 8*(L>=16).
__device__ __forceinline__ v16h load_a(const _Float16* __restrict__ src, int mbase,
                                       int stride, int k0) {
    int l = lane_id(), hi = l >> 4, m = l & 15;
    const _Float16* p = src + (size_t)(mbase + m) * stride + k0 + 8 * hi;
    v8h lo = *(const v8h*)p;
    v8h hh = *(const v8h*)(p + 16);
    return make16(lo, hh);
}

// B-operand (32x16 f16, K x N), source layout [N][K] with stride `stride` halves.
// ISA layout: lane L holds column N = L&15; half e holds K = k0 + (e<8 ? e : e+8) + 8*(L>=16).
__device__ __forceinline__ v16h load_b(const _Float16* src, int nbase,
                                       int stride, int k0) {
    int l = lane_id(), hi = l >> 4, n = l & 15;
    const _Float16* p = src + (size_t)(nbase + n) * stride + k0 + 8 * hi;
    v8h lo = *(const v8h*)p;
    v8h hh = *(const v8h*)(p + 16);
    return make16(lo, hh);
}

__device__ __forceinline__ v8f wmma16(v16h a, v16h b, v8f c) {
    return __builtin_amdgcn_wmma_f32_16x16x32_f16(false, a, false, b, (short)0, c,
                                                  false, false);
}

__device__ __forceinline__ void wave_ds_fence() {
    asm volatile("s_wait_dscnt 0" ::: "memory");
    __builtin_amdgcn_wave_barrier();
}

// CDNA5 async global->LDS copy (GVS form): lds_off is the byte offset inside the
// workgroup's LDS allocation, voff the per-lane
// byte offset added to the 64-bit SGPR base. Tracked by ASYNCcnt.
__device__ __forceinline__ void async_ld16(unsigned lds_off, const void* base,
                                           unsigned voff) {
    asm volatile("global_load_async_to_lds_b128 %0, %1, %2"
                 :: "v"(lds_off), "v"(voff), "s"(base)
                 : "memory");
}

__device__ __forceinline__ void async_wait0() {
    asm volatile("s_wait_asynccnt 0" ::: "memory");
}

// ---------------- GroupNorm statistics: one block per (b, g) ----------------
__global__ void k_gn_stats(const float* __restrict__ x, float2* __restrict__ stats) {
    int bg = blockIdx.x;                       // b*32 + g ; group data is contiguous
    const float* p = x + (size_t)bg * (CPG * NPIX);
    float s = 0.f, s2 = 0.f;
    for (int i = threadIdx.x; i < CPG * NPIX; i += 256) {
        float v = p[i];
        s += v; s2 += v * v;
    }
    __shared__ float rs[256], rq[256];
    rs[threadIdx.x] = s; rq[threadIdx.x] = s2;
    __syncthreads();
    for (int o = 128; o > 0; o >>= 1) {
        if ((int)threadIdx.x < o) {
            rs[threadIdx.x] += rs[threadIdx.x + o];
            rq[threadIdx.x] += rq[threadIdx.x + o];
        }
        __syncthreads();
    }
    if (threadIdx.x == 0) {
        const float inv = 1.0f / (float)(CPG * NPIX);
        float m = rs[0] * inv;
        float var = rq[0] * inv - m * m;
        stats[bg] = make_float2(m, rsqrtf(var + 1e-6f));
    }
}

// ---------------- f32 -> f16 weight conversion ----------------
__global__ void k_cvt(const float* __restrict__ s, _Float16* __restrict__ d, int n) {
    int i = blockIdx.x * 256 + threadIdx.x;
    if (i < n) d[i] = (_Float16)s[i];
}

// -------- GroupNorm apply + transpose:  x[B][C][N] f32 -> xnT[B][N][C] f16 --------
__global__ void k_normT(const float* __restrict__ x, const float* __restrict__ gw,
                        const float* __restrict__ gb, const float2* __restrict__ stats,
                        _Float16* __restrict__ xnT) {
    __shared__ _Float16 tile[32][33];
    int b = blockIdx.z, c0 = blockIdx.y * 32, i0 = blockIdx.x * 32;
    int tx = threadIdx.x & 31, ty = threadIdx.x >> 5;
    for (int cc = ty; cc < 32; cc += 8) {
        int c = c0 + cc;
        float2 st = stats[b * GROUPS + (c >> 3)];
        float v = x[((size_t)b * CH + c) * NPIX + i0 + tx];
        v = (v - st.x) * st.y * gw[c] + gb[c];
        tile[cc][tx] = (_Float16)v;
    }
    __syncthreads();
    for (int ii = ty; ii < 32; ii += 8)
        xnT[((size_t)b * NPIX + i0 + ii) * CH + c0 + tx] = tile[tx][ii];
}

// ---------------- QKV projection GEMM: y[o,i] = W[o,:] . xn[:,i] + bias ----------------
// mode 0: store transposed, scaled by C^-0.5 (qT) ; mode 1: store transposed (kT)
// mode 2: store [C][N] direct (v)
__global__ void k_gemm_qkv(const _Float16* __restrict__ w16, const float* __restrict__ bias,
                           const _Float16* __restrict__ xnT, _Float16* __restrict__ dst,
                           int mode) {
    int b = blockIdx.z;
    int mt = blockIdx.y * 16;
    int wv = threadIdx.x >> 5;
    int n0 = blockIdx.x * 512 + wv * 64;                // wave tile: 16(M) x 64(N)
    const _Float16* bsrc = xnT + (size_t)b * NPIX * CH;

    v8f acc[4];
#pragma unroll
    for (int t = 0; t < 4; ++t) acc[t] = zero8();

    for (int k = 0; k < CH; k += 32) {
        v16h a = load_a(w16, mt, CH, k);
#pragma unroll
        for (int nt = 0; nt < 4; ++nt) {
            v16h bb = load_b(bsrc, n0 + nt * 16, CH, k);
            acc[nt] = wmma16(a, bb, acc[nt]);
        }
    }

    int l = lane_id(), hi = l >> 4, n = l & 15;
    int mb = mt + 8 * hi;
    float bv[8];
#pragma unroll
    for (int r = 0; r < 8; ++r) bv[r] = bias[mb + r];

    if (mode < 2) {
        float scale = (mode == 0) ? 0.0625f : 1.0f;     // fold C^-0.5 into q
#pragma unroll
        for (int nt = 0; nt < 4; ++nt) {
            v8h h8;
#pragma unroll
            for (int r = 0; r < 8; ++r)
                h8[r] = (_Float16)((acc[nt][r] + bv[r]) * scale);
            int i = n0 + nt * 16 + n;
            *(v8h*)(dst + ((size_t)b * NPIX + i) * CH + mb) = h8;
        }
    } else {
#pragma unroll
        for (int nt = 0; nt < 4; ++nt) {
            int i = n0 + nt * 16 + n;
#pragma unroll
            for (int r = 0; r < 8; ++r)
                dst[((size_t)b * CH + mb + r) * NPIX + i] = (_Float16)(acc[nt][r] + bv[r]);
        }
    }
}

// ---------------- Flash attention: WG = 128 queries (8 waves x 16 rows) ----------------
// Per 32-key chunk the whole WG stages K (32x256 f16) and V (256x32 f16) into LDS
// with async global->LDS copies (all 8 waves consume the same B operands).
#define KS_STRIDE 264                 // 256 + 8 halves pad (bank spread)
#define VS_STRIDE 40                  // 32 + 8 halves pad
#define KS_OFF    0
#define VS_OFF    (32 * KS_STRIDE)                       // 8448
#define PS_OFF    (VS_OFF + 256 * VS_STRIDE)             // 18688
#define PS_STRIDE 40
#define SMEM_HALVES (PS_OFF + 8 * 16 * PS_STRIDE)        // 23808 halves = 47616 B

__global__ void __launch_bounds__(256, 1)
k_attn(const _Float16* __restrict__ qT, const _Float16* __restrict__ kT,
       const _Float16* __restrict__ vv, _Float16* __restrict__ attnT) {
    __shared__ _Float16 smem[SMEM_HALVES];
    int b = blockIdx.y, i0 = blockIdx.x * 128;
    int wv = threadIdx.x >> 5;
    int l = lane_id(), hi = l >> 4, nn = l & 15;

    const _Float16* qb = qT + (size_t)b * NPIX * CH;
    const _Float16* kb = kT + (size_t)b * NPIX * CH;
    const _Float16* vb = vv + (size_t)b * CH * NPIX;
    int irow = i0 + wv * 16;
    _Float16* myP = &smem[PS_OFF + wv * 16 * PS_STRIDE];

    v8f o[16];
#pragma unroll
    for (int t = 0; t < 16; ++t) o[t] = zero8();
    float mrow[8], lrow[8];
#pragma unroll
    for (int r = 0; r < 8; ++r) { mrow[r] = -1e30f; lrow[r] = 0.f; }

    for (int jc = 0; jc < NPIX; jc += 32) {
        // ---- stage K chunk [32][256] and V chunk [256][32] via async copies ----
        __syncthreads();                                 // previous chunk fully consumed
        {
            int t = threadIdx.x;
#pragma unroll
            for (int q = 0; q < 4; ++q) {                // K: 1024 x 16B chunks
                int idx = t + q * 256;
                int row = idx >> 5, ch = idx & 31;
                unsigned ldso = (unsigned)((KS_OFF + row * KS_STRIDE + ch * 8) * 2);
                unsigned goff = (unsigned)(((jc + row) * CH + ch * 8) * 2);
                async_ld16(ldso, kb, goff);
            }
#pragma unroll
            for (int q = 0; q < 4; ++q) {                // V: 256 rows x 4 chunks
                int idx = t + q * 256;
                int c = idx >> 2, ch = idx & 3;
                unsigned ldso = (unsigned)((VS_OFF + c * VS_STRIDE + ch * 8) * 2);
                unsigned goff = (unsigned)((c * NPIX + jc + ch * 8) * 2);
                async_ld16(ldso, vb, goff);
            }
            async_wait0();
        }
        __syncthreads();                                 // chunk visible to all waves

        // ---- S = (q * C^-0.5)^T k : two 16x16 j-tiles, K-loop over C ----
        v8f s0 = zero8(), s1 = zero8();
        for (int k = 0; k < CH; k += 32) {
            v16h a  = load_a(qb, irow, CH, k);           // per-wave rows, L0-resident
            v16h b0 = load_b(&smem[KS_OFF], 0,  KS_STRIDE, k);
            v16h b1 = load_b(&smem[KS_OFF], 16, KS_STRIDE, k);
            s0 = wmma16(a, b0, s0);
            s1 = wmma16(a, b1, s1);
        }
        // ---- online softmax across the 32 new columns ----
        float alpha[8];
#pragma unroll
        for (int r = 0; r < 8; ++r) {
            float rm = fmaxf(s0[r], s1[r]);
            rm = fmaxf(rm, __shfl_xor(rm, 1, 32));
            rm = fmaxf(rm, __shfl_xor(rm, 2, 32));
            rm = fmaxf(rm, __shfl_xor(rm, 4, 32));
            rm = fmaxf(rm, __shfl_xor(rm, 8, 32));
            float mn = fmaxf(mrow[r], rm);
            alpha[r] = __expf(mrow[r] - mn);
            mrow[r] = mn;
            float p0 = __expf(s0[r] - mn);
            float p1 = __expf(s1[r] - mn);
            int row = r + 8 * hi;                        // C-layout -> P row index
            myP[row * PS_STRIDE + nn]      = (_Float16)p0;
            myP[row * PS_STRIDE + 16 + nn] = (_Float16)p1;
            float rs = p0 + p1;
            rs += __shfl_xor(rs, 1, 32);
            rs += __shfl_xor(rs, 2, 32);
            rs += __shfl_xor(rs, 4, 32);
            rs += __shfl_xor(rs, 8, 32);
            lrow[r] = lrow[r] * alpha[r] + rs;
        }
#pragma unroll
        for (int t = 0; t < 16; ++t)
#pragma unroll
            for (int r = 0; r < 8; ++r) o[t][r] *= alpha[r];

        wave_ds_fence();                                 // P stores -> P loads (same wave)

        // ---- O += P @ V^T : A = P (16x32 from LDS), 16 c-tiles of B from LDS V ----
        const _Float16* pp = myP + nn * PS_STRIDE + 8 * hi;
        v8h plo = *(const v8h*)pp;
        v8h phi = *(const v8h*)(pp + 16);
        v16h pa = make16(plo, phi);
#pragma unroll
        for (int ct = 0; ct < 16; ++ct) {
            v16h bb = load_b(&smem[VS_OFF], ct * 16, VS_STRIDE, 0);
            o[ct] = wmma16(pa, bb, o[ct]);
        }
        __builtin_amdgcn_wave_barrier();                 // keep next-iter stores behind loads
    }

    float inv[8];
#pragma unroll
    for (int r = 0; r < 8; ++r) inv[r] = (lrow[r] > 0.f) ? (1.0f / lrow[r]) : 0.f;

    _Float16* ab = attnT + (size_t)b * NPIX * CH;        // attnT[B][N][C] f16
#pragma unroll
    for (int ct = 0; ct < 16; ++ct)
#pragma unroll
        for (int r = 0; r < 8; ++r) {
            int i = irow + r + 8 * hi;
            ab[(size_t)i * CH + ct * 16 + nn] = (_Float16)(o[ct][r] * inv[r]);
        }
}

// ---------------- Output projection + bias + residual ----------------
__global__ void k_gemm_out(const _Float16* __restrict__ wo16, const float* __restrict__ bo,
                           const _Float16* __restrict__ attnT, const float* __restrict__ x,
                           float* __restrict__ out) {
    int b = blockIdx.z;
    int mt = blockIdx.y * 16;
    int wv = threadIdx.x >> 5;
    int n0 = blockIdx.x * 512 + wv * 64;
    const _Float16* bsrc = attnT + (size_t)b * NPIX * CH;

    v8f acc[4];
#pragma unroll
    for (int t = 0; t < 4; ++t) acc[t] = zero8();

    for (int k = 0; k < CH; k += 32) {
        v16h a = load_a(wo16, mt, CH, k);
#pragma unroll
        for (int nt = 0; nt < 4; ++nt) {
            v16h bb = load_b(bsrc, n0 + nt * 16, CH, k);
            acc[nt] = wmma16(a, bb, acc[nt]);
        }
    }

    int l = lane_id(), hi = l >> 4, n = l & 15;
    int mb = mt + 8 * hi;
    float bv[8];
#pragma unroll
    for (int r = 0; r < 8; ++r) bv[r] = bo[mb + r];
    const float* xb = x + (size_t)b * CH * NPIX;
    float* ob = out + (size_t)b * CH * NPIX;
#pragma unroll
    for (int nt = 0; nt < 4; ++nt) {
        int i = n0 + nt * 16 + n;
#pragma unroll
        for (int r = 0; r < 8; ++r) {
            size_t idx = (size_t)(mb + r) * NPIX + i;
            ob[idx] = acc[nt][r] + bv[r] + xb[idx];
        }
    }
}

// ---------------- host-side orchestration ----------------
extern "C" void kernel_launch(void* const* d_in, const int* in_sizes, int n_in,
                              void* d_out, int out_size, void* d_ws, size_t ws_size,
                              hipStream_t stream) {
    (void)in_sizes; (void)n_in; (void)out_size; (void)ws_size;
    const float* x    = (const float*)d_in[0];
    const float* gn_w = (const float*)d_in[1];
    const float* gn_b = (const float*)d_in[2];
    const float* wq   = (const float*)d_in[3];
    const float* bq   = (const float*)d_in[4];
    const float* wk   = (const float*)d_in[5];
    const float* bk   = (const float*)d_in[6];
    const float* wvp  = (const float*)d_in[7];
    const float* bvp  = (const float*)d_in[8];
    const float* wo   = (const float*)d_in[9];
    const float* bo   = (const float*)d_in[10];
    float* out = (float*)d_out;

    char* ws = (char*)d_ws;
    const size_t WMAT = (size_t)CH * CH;                         // 65536 elements
    const size_t STATS_OFF = 0;
    const size_t W16_OFF   = 4096;
    const size_t XNT_OFF   = 1u << 20;                           // 1 MB
    const size_t PLANE16   = (size_t)BATCH * NPIX * CH * sizeof(_Float16); // 16 MB
    const size_t QT_OFF    = XNT_OFF + PLANE16;
    const size_t KT_OFF    = QT_OFF + PLANE16;
    const size_t V_OFF     = KT_OFF + PLANE16;
    const size_t AT_OFF    = V_OFF + PLANE16;

    float2*    stats = (float2*)(ws + STATS_OFF);
    _Float16*  wq16  = (_Float16*)(ws + W16_OFF);
    _Float16*  wk16  = wq16 + WMAT;
    _Float16*  wv16  = wk16 + WMAT;
    _Float16*  wo16  = wv16 + WMAT;
    _Float16*  xnT   = (_Float16*)(ws + XNT_OFF);
    _Float16*  qT    = (_Float16*)(ws + QT_OFF);
    _Float16*  kT    = (_Float16*)(ws + KT_OFF);
    _Float16*  vbuf  = (_Float16*)(ws + V_OFF);
    _Float16*  aT    = (_Float16*)(ws + AT_OFF);

    k_gn_stats<<<BATCH * GROUPS, 256, 0, stream>>>(x, stats);

    int wblocks = (int)((WMAT + 255) / 256);
    k_cvt<<<wblocks, 256, 0, stream>>>(wq,  wq16, (int)WMAT);
    k_cvt<<<wblocks, 256, 0, stream>>>(wk,  wk16, (int)WMAT);
    k_cvt<<<wblocks, 256, 0, stream>>>(wvp, wv16, (int)WMAT);
    k_cvt<<<wblocks, 256, 0, stream>>>(wo,  wo16, (int)WMAT);

    k_normT<<<dim3(NPIX / 32, CH / 32, BATCH), 256, 0, stream>>>(x, gn_w, gn_b, stats, xnT);

    dim3 ggrid(NPIX / 512, CH / 16, BATCH);
    k_gemm_qkv<<<ggrid, 256, 0, stream>>>(wq16, bq,  xnT, qT,   0);
    k_gemm_qkv<<<ggrid, 256, 0, stream>>>(wk16, bk,  xnT, kT,   1);
    k_gemm_qkv<<<ggrid, 256, 0, stream>>>(wv16, bvp, xnT, vbuf, 2);

    k_attn<<<dim3(NPIX / 128, BATCH), 256, 0, stream>>>(qT, kT, vbuf, aT);

    k_gemm_out<<<ggrid, 256, 0, stream>>>(wo16, bo, aT, x, out);
}
